// HybridAttention_64261300683343
// MI455X (gfx1250) — matrix operand order repaired and checked by reading, not compiled, not run on hardware
//
#include <hip/hip_runtime.h>
#include <math.h>

typedef __bf16 bf16_t;
typedef bf16_t v16bf __attribute__((ext_vector_type(16)));
typedef bf16_t v8bf  __attribute__((ext_vector_type(8)));
typedef bf16_t v4bf  __attribute__((ext_vector_type(4)));
typedef float  v8f   __attribute__((ext_vector_type(8)));
typedef float  v4f   __attribute__((ext_vector_type(4)));

#define B_  2
#define S_  2048
#define H_  16
#define D_  128
#define BM  128    // query rows per block
#define BN  64     // keys per kv tile
#define NW  8      // waves per block
#define NT  256    // threads per block
#define VROWS (D_ + 16)   // vt rows: 128 data + ones row (l-column) + 15 zero rows

__launch_bounds__(NT)
__global__ void fa_fwd_causal(const float* __restrict__ Q,
                              const float* __restrict__ K,
                              const float* __restrict__ V,
                              float* __restrict__ O)
{
    __shared__ bf16_t qs[BM * D_];        // [row][d]               32 KB
    __shared__ bf16_t ks[BN * D_];        // [kv][d]                16 KB
    __shared__ bf16_t vt[VROWS * BN];     // [d][kv] transposed     18 KB
    __shared__ bf16_t ps[NW * 16 * BN];   // per-wave P stage       16 KB

    const int tid  = threadIdx.x;
    const int lane = tid & 31;
    const int wave = tid >> 5;
    const int lh   = lane & 15;   // lane within half-wave
    const int hi   = lane >> 4;   // 0: lanes 0-15, 1: lanes 16-31

    const int m0 = blockIdx.x * BM;
    const int b  = blockIdx.y / H_;
    const int h  = blockIdx.y % H_;

    // 1/sqrt(128) * log2(e): scores go straight to the exp2 domain
    const float sscale = 0.08838834764831845f * 1.4426950408889634f;

    // ---- stage Q tile (float4 loads, packed bf16 stores) ----
    {
        const float* qb = Q + (((size_t)b * S_ + m0) * H_ + h) * D_;
        for (int i = 0; i < (BM * D_ / 4) / NT; ++i) {
            int e = i * NT + tid;          // float4 index
            int r = e >> 5;                // 32 float4 per row
            int d = (e & 31) * 4;
            v4f x = *(const v4f*)(qb + (size_t)r * H_ * D_ + d);
            v4bf y = { (bf16_t)x[0], (bf16_t)x[1], (bf16_t)x[2], (bf16_t)x[3] };
            *(v4bf*)&qs[r * D_ + d] = y;
        }
        // l-column rows of vt: row 128 = ones, rows 129..143 = zeros (written once;
        // tile staging only touches rows 0..127)
        for (int e = tid; e < 16 * BN; e += NT) {
            int rr = e >> 6;
            vt[(D_ + rr) * BN + (e & 63)] = (bf16_t)(rr == 0 ? 1.0f : 0.0f);
        }
    }

    const int wrow = wave * 16;

    float mrow[8];
    v8f   oacc[8], lacc;   // O tile 16x128 + l column (ones-trick accumulator)
    #pragma unroll
    for (int r = 0; r < 8; ++r) mrow[r] = -INFINITY;
    #pragma unroll
    for (int c = 0; c < 8; ++c) oacc[c] = (v8f)0.0f;
    lacc = (v8f)0.0f;

    const int ntiles = m0 / BN + 2;   // causal: kv tiles cover 0 .. m0+BM-1

    for (int j = 0; j < ntiles; ++j) {
        __syncthreads();
        // ---- stage K (row-major) and V (transposed) tiles ----
        {
            const float* kb = K + (((size_t)b * S_ + j * BN) * H_ + h) * D_;
            const float* vb = V + (((size_t)b * S_ + j * BN) * H_ + h) * D_;
            for (int i = 0; i < (BN * D_ / 4) / NT; ++i) {
                int e  = i * NT + tid;
                int kv = e >> 5;
                int d  = (e & 31) * 4;
                size_t g = (size_t)kv * H_ * D_ + d;
                v4f xk = *(const v4f*)(kb + g);
                v4f xv = *(const v4f*)(vb + g);
                v4bf yk = { (bf16_t)xk[0], (bf16_t)xk[1], (bf16_t)xk[2], (bf16_t)xk[3] };
                *(v4bf*)&ks[kv * D_ + d] = yk;
                #pragma unroll
                for (int t = 0; t < 4; ++t)
                    vt[(d + t) * BN + kv] = (bf16_t)xv[t];
            }
        }
        __syncthreads();

        // ---- S = Q(16x128) . K^T(128x64): four 16x16 f32 accumulators ----
        v8f s[4];
        #pragma unroll
        for (int c = 0; c < 4; ++c) s[c] = (v8f)0.0f;
        #pragma unroll
        for (int d0 = 0; d0 < D_; d0 += 32) {
            v16bf a;
            {
                const bf16_t* p = &qs[(wrow + lh) * D_ + d0 + hi * 16];
                v8bf lo = *(const v8bf*)p;
                v8bf hv = *(const v8bf*)(p + 8);
                #pragma unroll
                for (int i = 0; i < 8; ++i) { a[i] = lo[i]; a[i + 8] = hv[i]; }
            }
            #pragma unroll
            for (int c = 0; c < 4; ++c) {
                v16bf bk = *(const v16bf*)&ks[(c * 16 + lh) * D_ + d0 + hi * 16];
                s[c] = __builtin_amdgcn_wmma_f32_16x16x32_bf16(false, a, false, bk,
                                                               (short)0, s[c],
                                                               false, false);
            }
        }

        // ---- scale into exp2 domain; mask only near the diagonal (uniform branch) ----
        float sv[4][8];
        #pragma unroll
        for (int c = 0; c < 4; ++c)
            #pragma unroll
            for (int r = 0; r < 8; ++r)
                sv[c][r] = s[c][r] * sscale;
        if (j >= ntiles - 2) {
            #pragma unroll
            for (int c = 0; c < 4; ++c) {
                int col = j * BN + c * 16 + lh;
                #pragma unroll
                for (int r = 0; r < 8; ++r) {
                    int row = m0 + wrow + r + hi * 8;
                    if (col > row) sv[c][r] = -INFINITY;
                }
            }
        }

        // ---- online max + exp2; P written straight into per-wave LDS stage ----
        bf16_t* pw = &ps[wave * 16 * BN];
        float alpha[8];
        #pragma unroll
        for (int r = 0; r < 8; ++r) {
            float mx = fmaxf(fmaxf(sv[0][r], sv[1][r]), fmaxf(sv[2][r], sv[3][r]));
            #pragma unroll
            for (int off = 1; off < 16; off <<= 1)
                mx = fmaxf(mx, __shfl_xor(mx, off, 16));
            float mnew = fmaxf(mrow[r], mx);
            alpha[r] = __builtin_amdgcn_exp2f(mrow[r] - mnew);
            mrow[r] = mnew;
            int row = r + hi * 8;
            #pragma unroll
            for (int c = 0; c < 4; ++c)
                pw[row * BN + c * 16 + lh] =
                    (bf16_t)__builtin_amdgcn_exp2f(sv[c][r] - mnew);
        }

        // ---- rescale running O and l (row-sum lives in lacc via ones column) ----
        #pragma unroll
        for (int c = 0; c < 8; ++c)
            #pragma unroll
            for (int r = 0; r < 8; ++r)
                oacc[c][r] *= alpha[r];
        #pragma unroll
        for (int r = 0; r < 8; ++r) lacc[r] *= alpha[r];

        asm volatile("s_wait_dscnt 0x0" ::: "memory");  // within-wave cross-lane LDS dep

        // ---- P A-fragments (two K=32 chunks of the 16x64 P tile) ----
        v16bf pa[2];
        #pragma unroll
        for (int c = 0; c < 2; ++c) {
            const bf16_t* p = &ps[wave * 16 * BN + lh * BN + c * 32 + hi * 16];
            v8bf lo = *(const v8bf*)p;
            v8bf hv = *(const v8bf*)(p + 8);
            #pragma unroll
            for (int i = 0; i < 8; ++i) { pa[c][i] = lo[i]; pa[c][i + 8] = hv[i]; }
        }

        // ---- O(16x128) += P(16x64) . V(64x128); dc==8 accumulates l via ones ----
        #pragma unroll
        for (int dc = 0; dc < 9; ++dc) {
            #pragma unroll
            for (int c = 0; c < 2; ++c) {
                v16bf vb = *(const v16bf*)&vt[(dc * 16 + lh) * BN + c * 32 + hi * 16];
                v8f acc = (dc < 8) ? oacc[dc] : lacc;
                acc = __builtin_amdgcn_wmma_f32_16x16x32_bf16(false, pa[c], false, vb,
                                                              (short)0, acc,
                                                              false, false);
                if (dc < 8) oacc[dc] = acc; else lacc = acc;
            }
        }
    }

    // ---- epilogue: broadcast l (column 0 of the dc==8 tile), divide, store ----
    {
        float* ob = O + (((size_t)b * S_ + m0 + wrow) * H_ + h) * D_;
        #pragma unroll
        for (int r = 0; r < 8; ++r) {
            int   row = r + hi * 8;
            float l   = __shfl(lacc[r], hi * 16, 32);
            float inv = __builtin_amdgcn_rcpf(l);
            #pragma unroll
            for (int c = 0; c < 8; ++c)
                ob[(size_t)row * H_ * D_ + c * 16 + lh] = oacc[c][r] * inv;
        }
    }
}

extern "C" void kernel_launch(void* const* d_in, const int* in_sizes, int n_in,
                              void* d_out, int out_size, void* d_ws, size_t ws_size,
                              hipStream_t stream) {
    const float* Q = (const float*)d_in[0];
    const float* K = (const float*)d_in[1];
    const float* V = (const float*)d_in[2];
    float*       O = (float*)d_out;
    dim3 grid(S_ / BM, B_ * H_);
    fa_fwd_causal<<<grid, NT, 0, stream>>>(Q, K, V, O);
}
